// GraphAttentionLayer_52871047413896
// MI455X (gfx1250) — compile-verified, matrix operand order checked by reading, not run on hardware
//
#include <hip/hip_runtime.h>
#include <math.h>

// ---------------------------------------------------------------- constants
#define NNODES 20000
#define NEDGES 640000
#define IN_F   256
#define OUT_F  64
#define NHEADS 8
#define BATCH  2
#define ALPHA_SLOPE 0.2f

typedef __attribute__((ext_vector_type(16))) __bf16 v16bf;
typedef __attribute__((ext_vector_type(8)))  float  v8f;

// round-to-nearest-even f32 -> bf16 (used only in one-time prep kernels)
static __device__ __forceinline__ __bf16 f2bf(float f) {
    union { float f; unsigned u; } v; v.f = f;
    unsigned r = v.u + 0x7FFFu + ((v.u >> 16) & 1u);
    unsigned short h = (unsigned short)(r >> 16);
    __bf16 out;
    __builtin_memcpy(&out, &h, 2);
    return out;
}

union FragCast { uint4 q[2]; v16bf v; };

// ------------------------------------------------------------------ zeroing
__global__ void gat_zero_f32(float* __restrict__ p, int n) {
    int i = blockIdx.x * blockDim.x + threadIdx.x;
    if (i < n) p[i] = 0.0f;
}

// ------------------------------------------- one-time: x (f32) -> xb (bf16)
__global__ void gat_cvt_x(const float* __restrict__ x, __bf16* __restrict__ xb, int n) {
    int i = blockIdx.x * blockDim.x + threadIdx.x;
    if (i < n) xb[i] = f2bf(x[i]);
}

// ---- one-time: pack W into per-lane B-fragment layout (bf16)
// Wpk[(((head*8 + kk)*64 + col)*2 + khalf)*16 + j]
//   j 0..7  -> K = kk*32 + khalf*8 + j
//   j 8..15 -> K = kk*32 + 16 + khalf*8 + (j-8)
__global__ void gat_pack_w(const float* __restrict__ W, __bf16* __restrict__ Wpk) {
    int o = blockIdx.x * blockDim.x + threadIdx.x;
    if (o >= NHEADS * 8 * 64 * 2 * 16) return;
    int j     = o & 15;
    int khalf = (o >> 4) & 1;
    int col   = (o >> 5) & 63;
    int kk    = (o >> 11) & 7;
    int head  = o >> 14;
    int K = kk * 32 + khalf * 8 + (j < 8 ? j : 8 + j);   // j>=8: +16 + (j-8) == 8+j
    Wpk[o] = f2bf(W[((size_t)head * IN_F + K) * OUT_F + col]);
}

// ---------------------------------------------------- GEMM: h = x @ W[head]
// One wave computes one 16x16 tile of h, K=256 via 8x v_wmma_f32_16x16x32_bf16.
// Block = 256 threads = 8 waves = 2 mtiles x 4 ntiles. grid = (625, BATCH).
// Packed per-head W (32KB bf16) is staged in LDS once per block.
__global__ void __launch_bounds__(256)
gat_gemm_wmma(const __bf16* __restrict__ xb,    // [B, N, IN_F] bf16
              const __bf16* __restrict__ Wpk,   // packed [8*64*2*16] bf16 (this head)
              float* __restrict__ h) {          // [B, N, OUT_F]
    __shared__ uint4 smem[2048];                // 32 KB

    const int tid  = threadIdx.x;
    const int wave = tid >> 5;
    const int lane = tid & 31;

    // cooperative stage of packed W into LDS (2048 uint4)
    const uint4* wsrc = (const uint4*)Wpk;
#pragma unroll
    for (int i = 0; i < 8; ++i) smem[tid + 256 * i] = wsrc[tid + 256 * i];
    __syncthreads();

    const int mtile = blockIdx.x * 2 + (wave >> 2);
    const int ntile = wave & 3;
    const int b     = blockIdx.y;

    const int row0  = mtile * 16;
    const int col   = ntile * 16 + (lane & 15);     // B / C / D column (0..63)
    const int khalf = lane >> 4;
    const int arow  = row0 + (lane & 15);

    const __bf16* xr = xb + ((size_t)b * NNODES + arow) * IN_F;

    v8f c = {};
#pragma unroll
    for (int kk = 0; kk < 8; ++kk) {
        const int k0 = kk * 32;
        if (kk < 7) __builtin_prefetch(xr + k0 + 32, 0, 1);  // global_prefetch_b8

        // A fragment: two contiguous 16B chunks of this row (already bf16)
        FragCast af;
        af.q[0] = *(const uint4*)(xr + k0 + khalf * 8);
        af.q[1] = *(const uint4*)(xr + k0 + 16 + khalf * 8);

        // B fragment: 32 contiguous bytes from LDS
        FragCast bf;
        const int fidx = ((kk * 64 + col) * 2 + khalf) * 2;  // uint4 index
        bf.q[0] = smem[fidx];
        bf.q[1] = smem[fidx + 1];

        // (neg_a, A, neg_b, B, c_mod, C, reuse_a, reuse_b)
        c = __builtin_amdgcn_wmma_f32_16x16x32_bf16(false, af.v, false, bf.v,
                                                    (short)0, c, false, false);
    }

    // C/D store: VGPR r -> row (row0 + khalf*8 + r), col.
    const int rowbase = row0 + khalf * 8;
    float* hp = h + ((size_t)b * NNODES + rowbase) * OUT_F + col;
#pragma unroll
    for (int r = 0; r < 8; ++r) hp[(size_t)r * OUT_F] = c[r];
}

// --------------------------------------------------------- e = h @ a[head]
__global__ void gat_e_kernel(const float* __restrict__ h,   // [B, N, OUT_F]
                             const float* __restrict__ ah,  // [OUT_F, 2] (this head)
                             float* __restrict__ e) {       // [B, N, 2]
    int i = blockIdx.x * blockDim.x + threadIdx.x;
    if (i >= BATCH * NNODES) return;
    const float* hp = h + (size_t)i * OUT_F;
    float e0 = 0.0f, e1 = 0.0f;
#pragma unroll 8
    for (int d = 0; d < OUT_F; ++d) {
        float v = hp[d];
        e0 += v * ah[2 * d];
        e1 += v * ah[2 * d + 1];
    }
    e[2 * i]     = e0;
    e[2 * i + 1] = e1;
}

static __device__ __forceinline__ float edge_weight(const float* __restrict__ e,
                                                    int b, int src, int dst) {
    float logit = e[((size_t)b * NNODES + src) * 2] +
                  e[((size_t)b * NNODES + dst) * 2 + 1];
    float l = logit > 0.0f ? logit : ALPHA_SLOPE * logit;   // leaky_relu
    l = fminf(50.0f, fmaxf(-50.0f, l));                     // clip
    return expf(-l);
}

// ---------------------------------------- pass 1: rowsum scatter-add (denom)
__global__ void gat_edge1(const float* __restrict__ e,      // [B, N, 2]
                          const int* __restrict__ ei,       // [2, E]
                          float* __restrict__ rowsum) {     // [B, N]
    int idx = blockIdx.x * blockDim.x + threadIdx.x;
    if (idx >= BATCH * NEDGES) return;
    int eid = idx % NEDGES;
    int b   = idx / NEDGES;
    int src = ei[eid];
    int dst = ei[NEDGES + eid];
    float w = edge_weight(e, b, src, dst);
    atomicAdd(&rowsum[(size_t)b * NNODES + src], w);
}

// --------------------- pass 2: message aggregation, one wave32 per (b, edge)
__global__ void gat_edge2(const float* __restrict__ h,      // [B, N, OUT_F]
                          const float* __restrict__ e,      // [B, N, 2]
                          const float* __restrict__ rowsum, // [B, N]
                          const int* __restrict__ ei,       // [2, E]
                          float* __restrict__ out,          // [B, N, NHEADS*OUT_F]
                          int head) {
    int wg = blockIdx.x * 8 + (threadIdx.x >> 5);
    if (wg >= BATCH * NEDGES) return;
    int lane = threadIdx.x & 31;
    int eid  = wg % NEDGES;
    int b    = wg / NEDGES;
    int src  = ei[eid];
    int dst  = ei[NEDGES + eid];

    float w     = edge_weight(e, b, src, dst);
    float alpha = w / (rowsum[(size_t)b * NNODES + src] + 1e-20f);

    const float* hd = h   + ((size_t)b * NNODES + dst) * OUT_F;
    float*       op = out + ((size_t)b * NNODES + src) * (NHEADS * OUT_F) + head * OUT_F;

    atomicAdd(&op[lane],      alpha * hd[lane]);
    atomicAdd(&op[lane + 32], alpha * hd[lane + 32]);
}

// ------------------------------------------------- final: NaN -> 0, then ELU
__global__ void gat_elu(float* __restrict__ out, int n) {
    int i = blockIdx.x * blockDim.x + threadIdx.x;
    if (i >= n) return;
    float v = out[i];
    if (v != v) v = 0.0f;                                   // isnan -> 0
    out[i] = v > 0.0f ? v : (expf(v) - 1.0f);               // elu (alpha=1)
}

// -------------------------------------------------------------------- launch
extern "C" void kernel_launch(void* const* d_in, const int* in_sizes, int n_in,
                              void* d_out, int out_size, void* d_ws, size_t ws_size,
                              hipStream_t stream) {
    const float* x  = (const float*)d_in[0];   // [B, N, IN_F]
    const float* W  = (const float*)d_in[1];   // [H, IN_F, OUT_F]
    const float* a  = (const float*)d_in[2];   // [H, OUT_F, 2]
    const int*   ei = (const int*)d_in[3];     // [2, E]
    float* out = (float*)d_out;                // [B, N, H*OUT_F]

    // workspace layout:
    //   h      [B*N*64]  f32   (10.24 MB)
    //   e      [B*N*2]   f32   ( 0.32 MB)
    //   rowsum [B*N]     f32   ( 0.16 MB)
    //   xb     [B*N*256] bf16  (20.48 MB)
    //   Wpk    [H*16384] bf16  ( 0.26 MB)
    float*  h_ws  = (float*)d_ws;
    float*  e_ws  = h_ws + (size_t)BATCH * NNODES * OUT_F;
    float*  rs_ws = e_ws + (size_t)BATCH * NNODES * 2;
    __bf16* xb    = (__bf16*)(rs_ws + (size_t)BATCH * NNODES);
    __bf16* Wpk   = xb + (size_t)BATCH * NNODES * IN_F;

    const int n_x   = BATCH * NNODES * IN_F;                 // 10,240,000
    const int n_wpk = NHEADS * 8 * 64 * 2 * 16;              // 131,072
    const int n_out = BATCH * NNODES * NHEADS * OUT_F;       // 20,480,000

    // one-time prep: bf16 mirror of x, packed bf16 W fragments
    gat_cvt_x<<<(n_x + 255) / 256, 256, 0, stream>>>(x, xb, n_x);
    gat_pack_w<<<(n_wpk + 255) / 256, 256, 0, stream>>>(W, Wpk);
    gat_zero_f32<<<(n_out + 255) / 256, 256, 0, stream>>>(out, n_out);

    for (int head = 0; head < NHEADS; ++head) {
        gat_gemm_wmma<<<dim3(NNODES / 32, BATCH), 256, 0, stream>>>(
            xb, Wpk + (size_t)head * 8 * 64 * 2 * 16, h_ws);

        gat_e_kernel<<<(BATCH * NNODES + 255) / 256, 256, 0, stream>>>(
            h_ws, a + (size_t)head * OUT_F * 2, e_ws);

        gat_zero_f32<<<(BATCH * NNODES + 255) / 256, 256, 0, stream>>>(
            rs_ws, BATCH * NNODES);

        gat_edge1<<<(BATCH * NEDGES + 255) / 256, 256, 0, stream>>>(
            e_ws, ei, rs_ws);

        gat_edge2<<<(BATCH * NEDGES + 7) / 8, 256, 0, stream>>>(
            h_ws, e_ws, rs_ws, ei, out, head);
    }

    gat_elu<<<(n_out + 255) / 256, 256, 0, stream>>>(out, n_out);
}
